// PCTransformer_29154238006045
// MI455X (gfx1250) — matrix operand order repaired and checked by reading, not compiled, not run on hardware
//
#include <hip/hip_runtime.h>
#include <cstdint>

// ---------------------------------------------------------------------------
// PCTransformer predictive-coding inference, MI455X / gfx1250.
//
// Exact dead-code elimination: the returned logits depend only on the closed
// recursion  eps = onehot(tgt) - (x_out@W_out + b);  x_out += g*(eps@W_out^T).
// The attention / FFN / LN state updates never feed back into x_out or the
// output, so they are skipped (bit-exact w.r.t. the reference dataflow).
//
// All GEMMs: v_wmma_f32_16x16x32_f16 (f16 inputs, f32 accum).
// W_out is converted once to f16 in both layouts (DxV and VxD) so every GEMM
// stages B as coalesced b128 loads of an N x K matrix.
// Staging uses gfx1250 async global->LDS copies (ASYNCcnt, double-buffered
// LDS, s_wait_asynccnt) — probe confirmed the builtin exists with signature
// (int4 AS1*, int4 AS3*, imm offset, imm cpol).  Register-prefetch fallback
// kept for toolchains without the builtin.
// ---------------------------------------------------------------------------

typedef _Float16 f16;
typedef __attribute__((ext_vector_type(16))) _Float16 v16h;
typedef __attribute__((ext_vector_type(8)))  float    v8f;

#define PC_B 2
#define PC_S 512
#define PC_D 512
#define PC_V 32000
#define PC_T 5
#define PC_GAMMA 0.1f

#define BM 128
#define BN 128
#define BK 32
#define TSTRIDE 40                 // halves per LDS tile row (80B: 16B-aligned rows,
                                   // 20-dword stride -> conflict-free frag reads)
#define TILE_H (128 * TSTRIDE)     // halves per staged tile (A or B)

#if __has_builtin(__builtin_amdgcn_global_load_async_to_lds_b128)
#define USE_ASYNC_LDS 1
typedef int v4i __attribute__((__vector_size__(16)));
typedef __attribute__((address_space(1))) v4i as1_v4i;   // global int4*
typedef __attribute__((address_space(3))) v4i as3_v4i;   // LDS int4*
#else
#define USE_ASYNC_LDS 0
#endif

__device__ __forceinline__ void wait_async_all() {
#if __has_builtin(__builtin_amdgcn_s_wait_asynccnt)
    __builtin_amdgcn_s_wait_asynccnt(0);
#else
    asm volatile("s_wait_asynccnt 0x0" ::: "memory");
#endif
}

enum { EPI_BIAS_F32 = 0, EPI_ONEHOT_F16 = 1, EPI_ATOMIC_F32 = 2 };

// ---------------------------------------------------------------------------
// Tiled WMMA GEMM:  C(M x N) = A(M x K) * B^T_stored, with B stored N x K
// (leading dim ldb over K).  Requires M%BM==0, N%BN==0, (K/gridDim.z)%BK==0,
// and 16B-aligned rows — true for every shape launched below.
//
// Epilogues (compile-time):
//   EPI_BIAS_F32  : Cf[r][c]  = acc + bias[c]                       (f32 out)
//   EPI_ONEHOT_F16: C16[r][c] = (ids[r]==c ? 1 : 0) - (acc+bias[c]) (f16 out)
//   EPI_ATOMIC_F32: atomicAdd(&Cf[r][c], acc)        (split-K over gridDim.z)
// ---------------------------------------------------------------------------
template <int EPI>
__global__ __launch_bounds__(256)
void gemm_wmma_f16(const f16* __restrict__ A,  long long lda,
                   const f16* __restrict__ Bt, long long ldb,
                   const float* __restrict__ bias,
                   const int*   __restrict__ ids,
                   float* __restrict__ Cf, long long ldc,
                   f16*   __restrict__ C16,
                   int K)
{
#if USE_ASYNC_LDS
    __shared__ f16 sMem[2 * 2 * TILE_H];   // [buf][A|B][...]
#else
    __shared__ f16 sMem[2 * TILE_H];       // [A|B][...]
#endif

    const int tid  = threadIdx.x;
    const int lane = tid & 31;
    const int wave = tid >> 5;
    const int wm   = (wave & 3) * 32;      // 4 waves tile M (32 rows each)
    const int wn   = (wave >> 2) * 64;     // 2 waves tile N (64 cols each)
    const int m_blk = blockIdx.y * BM;
    const int n_blk = blockIdx.x * BN;
    const int kper  = K / (int)gridDim.z;
    const int k_lo  = (int)blockIdx.z * kper;
    const int k_hi  = k_lo + kper;

    // staging geometry: 128 rows x 32 halves per tile, 16 halves / thread
    const int srow = tid >> 1;
    const int skb  = (tid & 1) * 16;
    const f16* gA = A  + (long long)(m_blk + srow) * lda + skb;
    const f16* gB = Bt + (long long)(n_blk + srow) * ldb + skb;

    v8f zero = {0.f, 0.f, 0.f, 0.f, 0.f, 0.f, 0.f, 0.f};
    v8f acc[2][4] = {{zero, zero, zero, zero}, {zero, zero, zero, zero}};

    // fragment loads for A(16x32)/B(32x16), both in contiguous k = 16g + s order, + WMMAs
    auto mma_tiles = [&](const f16* sA, const f16* sB) {
        const int g   = lane >> 4;
        const int l15 = lane & 15;
        union Frag { v16h v; unsigned int u[8]; };
        Frag fa[2], fb[4];
#pragma unroll
        for (int i = 0; i < 2; ++i) {
            const f16* base = sA + (wm + i * 16 + l15) * TSTRIDE;
#pragma unroll
            for (int p = 0; p < 4; ++p)       // VGPR0-3: K = 2p+16g, +1
                fa[i].u[p] = *(const unsigned int*)(base + 2 * p + 16 * g);
#pragma unroll
            for (int p = 0; p < 4; ++p)       // VGPR4-7: K = 8+2p+16g, +1
                fa[i].u[4 + p] = *(const unsigned int*)(base + 8 + 2 * p + 16 * g);
        }
#pragma unroll
        for (int j = 0; j < 4; ++j) {
            const f16* base = sB + (wn + j * 16 + l15) * TSTRIDE;
#pragma unroll
            for (int v = 0; v < 8; ++v)       // VGPRv: K = 2v+16g, +1
                fb[j].u[v] = *(const unsigned int*)(base + 2 * v + 16 * g);
        }
#pragma unroll
        for (int i = 0; i < 2; ++i)
#pragma unroll
            for (int j = 0; j < 4; ++j)
                acc[i][j] = __builtin_amdgcn_wmma_f32_16x16x32_f16(
                    false, fa[i].v, false, fb[j].v,
                    (short)0, acc[i][j], false, false);
    };

#if USE_ASYNC_LDS
    // ---- async global->LDS pipeline (ASYNCcnt), double-buffered ----
    auto stage_async = [&](f16* sA, f16* sB, int k0) {
        f16* dA = sA + srow * TSTRIDE + skb;
        f16* dB = sB + srow * TSTRIDE + skb;
        __builtin_amdgcn_global_load_async_to_lds_b128(
            (as1_v4i*)(gA + k0),     (as3_v4i*)dA,       0, 0);
        __builtin_amdgcn_global_load_async_to_lds_b128(
            (as1_v4i*)(gA + k0 + 8), (as3_v4i*)(dA + 8), 0, 0);
        __builtin_amdgcn_global_load_async_to_lds_b128(
            (as1_v4i*)(gB + k0),     (as3_v4i*)dB,       0, 0);
        __builtin_amdgcn_global_load_async_to_lds_b128(
            (as1_v4i*)(gB + k0 + 8), (as3_v4i*)(dB + 8), 0, 0);
    };
    int cur = 0;
    stage_async(sMem, sMem + TILE_H, k_lo);
    for (int k0 = k_lo; k0 < k_hi; k0 += BK) {
        wait_async_all();                                  // own tile landed
        __syncthreads();                                   // everyone's landed
        const int nxt = cur ^ 1;
        if (k0 + BK < k_hi)                                // overlap copy w/ WMMA
            stage_async(sMem + nxt * 2 * TILE_H,
                        sMem + nxt * 2 * TILE_H + TILE_H, k0 + BK);
        mma_tiles(sMem + cur * 2 * TILE_H,
                  sMem + cur * 2 * TILE_H + TILE_H);
        __syncthreads();
        cur = nxt;
    }
#else
    // ---- register-prefetch pipeline, single LDS buffer ----
    uint4 ra0, ra1, rb0, rb1;
    auto ld_regs = [&](int k0) {
        const uint4* pa = (const uint4*)(gA + k0); ra0 = pa[0]; ra1 = pa[1];
        const uint4* pb = (const uint4*)(gB + k0); rb0 = pb[0]; rb1 = pb[1];
    };
    ld_regs(k_lo);
    for (int k0 = k_lo; k0 < k_hi; k0 += BK) {
        uint4* dA = (uint4*)(sMem + srow * TSTRIDE + skb);
        uint4* dB = (uint4*)(sMem + TILE_H + srow * TSTRIDE + skb);
        dA[0] = ra0; dA[1] = ra1;
        dB[0] = rb0; dB[1] = rb1;
        __syncthreads();
        if (k0 + BK < k_hi) ld_regs(k0 + BK);   // issue next loads before WMMA
        mma_tiles(sMem, sMem + TILE_H);
        __syncthreads();
    }
#endif

    // ---- epilogue: 16x16 f32 C layout (lanes 0-15: N, VGPR r: M = r + 8g)
    const int g  = lane >> 4;
    const int cc = lane & 15;
#pragma unroll
    for (int i = 0; i < 2; ++i) {
#pragma unroll
        for (int j = 0; j < 4; ++j) {
            const int col = n_blk + wn + j * 16 + cc;
#pragma unroll
            for (int r = 0; r < 8; ++r) {
                const int row = m_blk + wm + i * 16 + 8 * g + r;
                const float val = acc[i][j][r];
                if (EPI == EPI_BIAS_F32) {
                    Cf[(long long)row * ldc + col] = val + bias[col];
                } else if (EPI == EPI_ONEHOT_F16) {
                    const float tgt = (ids[row] == col) ? 1.0f : 0.0f;
                    C16[(long long)row * ldc + col] = (f16)(tgt - (val + bias[col]));
                } else { // EPI_ATOMIC_F32 split-K reduction
                    atomicAdd(&Cf[(long long)row * ldc + col], val);
                }
            }
        }
    }
}

// x = 0, x16 = 0, g = 0
__global__ void pc_init(float* __restrict__ x, f16* __restrict__ x16,
                        float* __restrict__ g, int n)
{
    int i = blockIdx.x * blockDim.x + threadIdx.x;
    if (i < n) { x[i] = 0.f; x16[i] = (f16)0.f; g[i] = 0.f; }
}

// x += gamma*g ; x16 = (f16)x ; g = 0 (ready for next split-K GEMM)
__global__ void pc_update(float* __restrict__ x, f16* __restrict__ x16,
                          float* __restrict__ g, int n, float gamma)
{
    int i = blockIdx.x * blockDim.x + threadIdx.x;
    if (i < n) {
        float v = x[i] + gamma * g[i];
        x[i] = v;
        x16[i] = (f16)v;
        g[i] = 0.f;
    }
}

__global__ void cvt_f32_f16(const float* __restrict__ in, f16* __restrict__ out, int n)
{
    int i = blockIdx.x * blockDim.x + threadIdx.x;
    if (i < n) out[i] = (f16)in[i];
}

// out(C x R) = transpose(in(R x C)) with f32->f16 convert; R%32==0, C%32==0
__global__ __launch_bounds__(256)
void transpose_cvt_f32_f16(const float* __restrict__ in, f16* __restrict__ out,
                           int R, int C)
{
    __shared__ float tile[32][33];
    const int tx = threadIdx.x & 31;
    const int ty = threadIdx.x >> 5;            // 0..7
    const int r0 = blockIdx.y * 32;
    const int c0 = blockIdx.x * 32;
#pragma unroll
    for (int j = 0; j < 32; j += 8)
        tile[ty + j][tx] = in[(long long)(r0 + ty + j) * C + (c0 + tx)];
    __syncthreads();
#pragma unroll
    for (int j = 0; j < 32; j += 8)
        out[(long long)(c0 + ty + j) * R + (r0 + tx)] = (f16)tile[tx][ty + j];
}

extern "C" void kernel_launch(void* const* d_in, const int* in_sizes, int n_in,
                              void* d_out, int out_size, void* d_ws, size_t ws_size,
                              hipStream_t stream)
{
    (void)in_sizes; (void)n_in; (void)out_size;

    const int*   target_ids = (const int*)d_in[0];    // (B,S) flat, len 1024
    const float* W_out      = (const float*)d_in[20]; // (D,V) row-major f32
    const float* b_out      = (const float*)d_in[21]; // (V,)  f32

    const int M = PC_B * PC_S;                        // 1024 tokens

    // ---- workspace carve-out (256B-aligned offsets) ----
    char*  ws  = (char*)d_ws;
    size_t off = 0;
    auto carve = [&](size_t bytes) -> void* {
        void* p = ws + off;
        off += (bytes + 255) & ~(size_t)255;
        return p;
    };
    float* x_out = (float*)carve((size_t)M * PC_D * sizeof(float));
    float* gacc  = (float*)carve((size_t)M * PC_D * sizeof(float));
    f16*   x16   = (f16*)  carve((size_t)M * PC_D * sizeof(f16));
    f16*   W16   = (f16*)  carve((size_t)PC_D * PC_V * sizeof(f16)); // D x V
    f16*   W16T  = (f16*)  carve((size_t)PC_V * PC_D * sizeof(f16)); // V x D

    // eps (M x V f16, 65.5 MB): ws if it fits, else reuse d_out (131 MB f32)
    // as scratch — final GEMM overwrites d_out last and never reads eps.
    const size_t eps_bytes = (size_t)M * PC_V * sizeof(f16);
    f16* eps = (ws_size >= off + eps_bytes) ? (f16*)carve(eps_bytes)
                                            : (f16*)d_out;

    dim3 blk(256);

    // init latent state + split-K accumulator
    pc_init<<<(M * PC_D + 255) / 256, blk, 0, stream>>>(x_out, x16, gacc, M * PC_D);

    // W_out -> f16 in both layouts (one-time; 32 MB each, L2-resident after)
    const int nw = PC_D * PC_V;
    cvt_f32_f16<<<(nw + 255) / 256, blk, 0, stream>>>(W_out, W16, nw);
    transpose_cvt_f32_f16<<<dim3(PC_V / 32, PC_D / 32), blk, 0, stream>>>(
        W_out, W16T, PC_D, PC_V);

    for (int t = 0; t < PC_T; ++t) {
        // eps = onehot(target) - (x_out @ W_out + b_out)   [fused f16 out]
        dim3 g1(PC_V / BN, M / BM, 1);                 // (250, 8, 1)
        gemm_wmma_f16<EPI_ONEHOT_F16><<<g1, blk, 0, stream>>>(
            x16, PC_D, W16T, PC_D, b_out, target_ids,
            nullptr, PC_V, eps, /*K=*/PC_D);

        // gacc += eps @ W_out^T   [split-K=8, f32 atomics]
        dim3 g2(PC_D / BN, M / BM, 8);                 // (4, 8, 8)
        gemm_wmma_f16<EPI_ATOMIC_F32><<<g2, blk, 0, stream>>>(
            eps, PC_V, W16, PC_V, nullptr, nullptr,
            gacc, PC_D, nullptr, /*K=*/PC_V);

        // x_out += GAMMA * gacc ; refresh f16 copy ; rezero gacc
        pc_update<<<(M * PC_D + 255) / 256, blk, 0, stream>>>(
            x_out, x16, gacc, M * PC_D, PC_GAMMA);
    }

    // logits = x_out @ W_out + b_out  -> d_out (f32)
    dim3 g3(PC_V / BN, M / BM, 1);
    gemm_wmma_f16<EPI_BIAS_F32><<<g3, blk, 0, stream>>>(
        x16, PC_D, W16T, PC_D, b_out, nullptr,
        (float*)d_out, PC_V, nullptr, /*K=*/PC_D);
}
